// EvidenceGatedGDA_8160437863070
// MI455X (gfx1250) — matrix operand order repaired and checked by reading, not compiled, not run on hardware
//
#include <hip/hip_runtime.h>
#include <hip/hip_bf16.h>

typedef __attribute__((ext_vector_type(16))) _Float16 v16h;
typedef __attribute__((ext_vector_type(8)))  float    v8f;

#define B_    4
#define KBITS 8
#define NPIX  4096   // H*W = 64*64
#define DV    32
#define NCODE 256

// ---------------- Kernel 0: zero the histogram bins ----------------
__global__ void eg_init_counts(int* __restrict__ counts) {
    int t = blockIdx.x * blockDim.x + threadIdx.x;
    if (t < B_ * NCODE) counts[t] = 0;
}

// ---------------- Kernel 1: bitpack codes + evidence histogram ----------------
__global__ void eg_codes_hist(const float* __restrict__ z,
                              const int*   __restrict__ ev,
                              int* __restrict__ codes,
                              int* __restrict__ counts) {
    int t = blockIdx.x * blockDim.x + threadIdx.x;   // 0 .. B_*NPIX-1
    int b = t >> 12;
    int n = t & (NPIX - 1);
    int code = 0;
#pragma unroll
    for (int kk = 0; kk < KBITS; ++kk) {
        float zz = z[(b * KBITS + kk) * NPIX + n];
        code |= (zz > 0.5f) ? (1 << kk) : 0;
    }
    codes[t] = code;
    if (ev[t] != 0) atomicAdd(&counts[b * NCODE + code], 1);
}

// ---------------- Kernel 2: 256x256x48 GEMM per batch via WMMA ----------------
// D[q,d] = sum_c count[c]*exp(-popc(q^c)/temp) * B[c,d]
//   B[c,d] = value_embed[c][d] for d<32 ; 1.0 for d==32 (denominator) ; 0 pad
// One wave per (batch, row_tile of 16 q's). Split-f16: AhBh + AlBh + AhBl.
__global__ void __launch_bounds__(32)
eg_gemm_tables(const int*   __restrict__ counts,
               const float* __restrict__ ve,          // 256 x 32
               const float* __restrict__ temperature, // 1
               float* __restrict__ numer,             // B*256*32
               float* __restrict__ denom) {           // B*256
    const int b        = blockIdx.x >> 4;
    const int row_tile = blockIdx.x & 15;
    const int lane     = threadIdx.x;      // 0..31
    const int half     = lane >> 4;
    const int idx15    = lane & 15;
    const int q        = row_tile * 16 + idx15;   // A-fragment row for this lane
    const float inv_t  = 1.0f / fmaxf(temperature[0], 0.1f);

    v8f acc0 = {};   // d = 0..15
    v8f acc1 = {};   // d = 16..31
    v8f acc2 = {};   // d = 32 (softmax denominator), cols 33..47 pad

    for (int chunk = 0; chunk < 8; ++chunk) {       // K = 256 in steps of 32
        const int cbase = chunk * 32;
        v16h ah, al, b0h, b0l, b1h, b1l, b2h, b2l;
#pragma unroll
        for (int j = 0; j < 16; ++j) {
            // ---- A fragment: 16-bit A 16x32 per-lane K order: ----
            //   half 0 -> K = {0..7, 16..23} ; half 1 -> K = {8..15, 24..31}
            int kA = j + 8 * ((j >> 3) + half);
            int cA = cbase + kA;
            float cnt  = (float)counts[b * NCODE + cA];
            float w    = __expf(-(float)__popc(q ^ cA) * inv_t);
            float aval = cnt * w;
            _Float16 hh = (_Float16)aval;
            ah[j] = hh;
            al[j] = (_Float16)(aval - (float)hh);

            // ---- B fragments: 16-bit B 32x16 per-lane K order: ----
            //   half 0 -> K = {0..7, 16..23} ; half 1 -> K = {8..15, 24..31} ; column = idx15
            int cB = cbase + j + 8 * ((j >> 3) + half);
            float b0 = ve[cB * DV + idx15];            // d = idx15
            float b1 = ve[cB * DV + 16 + idx15];       // d = 16 + idx15
            float b2 = (idx15 == 0) ? 1.0f : 0.0f;     // d = 32 ones column
            _Float16 t0 = (_Float16)b0; b0h[j] = t0; b0l[j] = (_Float16)(b0 - (float)t0);
            _Float16 t1 = (_Float16)b1; b1h[j] = t1; b1l[j] = (_Float16)(b1 - (float)t1);
            b2h[j] = (_Float16)b2;      b2l[j] = (_Float16)0.0f;
        }
        // numerator tiles: 3-term split-precision accumulate
        acc0 = __builtin_amdgcn_wmma_f32_16x16x32_f16(false, ah, false, b0h, (short)0, acc0, false, false);
        acc0 = __builtin_amdgcn_wmma_f32_16x16x32_f16(false, al, false, b0h, (short)0, acc0, false, false);
        acc0 = __builtin_amdgcn_wmma_f32_16x16x32_f16(false, ah, false, b0l, (short)0, acc0, false, false);
        acc1 = __builtin_amdgcn_wmma_f32_16x16x32_f16(false, ah, false, b1h, (short)0, acc1, false, false);
        acc1 = __builtin_amdgcn_wmma_f32_16x16x32_f16(false, al, false, b1h, (short)0, acc1, false, false);
        acc1 = __builtin_amdgcn_wmma_f32_16x16x32_f16(false, ah, false, b1l, (short)0, acc1, false, false);
        // denominator: ones column is exact in f16, so only Ah+Al terms needed
        acc2 = __builtin_amdgcn_wmma_f32_16x16x32_f16(false, ah, false, b2h, (short)0, acc2, false, false);
        acc2 = __builtin_amdgcn_wmma_f32_16x16x32_f16(false, al, false, b2h, (short)0, acc2, false, false);
    }

    // C/D layout: lane<16 -> N=lane, VGPR r = row M=r ; lane>=16 -> M=r+8
    const int col   = idx15;
    const int rbase = half * 8;
#pragma unroll
    for (int r = 0; r < 8; ++r) {
        int qq = row_tile * 16 + rbase + r;
        numer[(b * NCODE + qq) * DV + col]      = acc0[r];
        numer[(b * NCODE + qq) * DV + 16 + col] = acc1[r];
        if (col == 0) denom[b * NCODE + qq]     = acc2[r];
    }
}

// ---------------- Kernel 3: per-pixel table lookup + divide ----------------
// out[((b*32+d)*4096)+n] = numer[b][code][d]/denom[b][code]  (mask_value if no evidence)
__global__ void eg_finalize(const int*   __restrict__ codes,
                            const float* __restrict__ numer,
                            const float* __restrict__ denom,
                            const float* __restrict__ mask_value,
                            float* __restrict__ out) {
    int t   = blockIdx.x * blockDim.x + threadIdx.x;  // 0 .. B_*DV*NPIX-1
    int b   = t >> 17;                 // / (DV*NPIX)
    int rem = t & (DV * NPIX - 1);
    int d   = rem >> 12;
    int n   = rem & (NPIX - 1);
    int q   = codes[b * NPIX + n];
    float den = denom[b * NCODE + q];
    out[t] = (den > 0.0f) ? (numer[(b * NCODE + q) * DV + d] / den) : mask_value[d];
}

extern "C" void kernel_launch(void* const* d_in, const int* in_sizes, int n_in,
                              void* d_out, int out_size, void* d_ws, size_t ws_size,
                              hipStream_t stream) {
    const float* z    = (const float*)d_in[0];   // (4, 8, 64, 64) f32
    const int*   ev   = (const int*)  d_in[1];   // (4, 4096) bool->int
    const float* ve   = (const float*)d_in[2];   // (256, 32) f32
    const float* mv   = (const float*)d_in[3];   // (32,) f32
    const float* temp = (const float*)d_in[4];   // (1,) f32
    float* out = (float*)d_out;                  // (4, 32, 64, 64) f32

    char* ws = (char*)d_ws;
    int*   codes  = (int*)  (ws);            // 16384 ints  (64 KB)
    int*   counts = (int*)  (ws + 65536);    // 1024 ints   (4 KB)
    float* denomp = (float*)(ws + 69632);    // 1024 f32    (4 KB)
    float* numerp = (float*)(ws + 73728);    // 32768 f32   (128 KB)

    eg_init_counts<<<4, 256, 0, stream>>>(counts);
    eg_codes_hist <<<(B_ * NPIX) / 256, 256, 0, stream>>>(z, ev, codes, counts);
    eg_gemm_tables<<<B_ * 16, 32, 0, stream>>>(counts, ve, temp, numerp, denomp);
    eg_finalize   <<<(B_ * DV * NPIX) / 256, 256, 0, stream>>>(codes, numerp, denomp, mv, out);
}
